// MoEMLABlock_86002425135555
// MI455X (gfx1250) — matrix operand order repaired and checked by reading, not compiled, not run on hardware
//
#include <hip/hip_runtime.h>
#include <math.h>

typedef __attribute__((ext_vector_type(16))) _Float16 v16h;
typedef __attribute__((ext_vector_type(8)))  _Float16 v8h;
typedef __attribute__((ext_vector_type(2)))  __fp16   h2;   // cvt_pkrtz result type
typedef __attribute__((ext_vector_type(8)))  float    v8f;
typedef __attribute__((ext_vector_type(4)))  int      v4i;

#define DEVINL __device__ __forceinline__

constexpr int Bc  = 2;
constexpr int Sc  = 2048;
constexpr int Hc  = 2048;
constexpr int NHc = 16;
constexpr int HDc = 128;
constexpr int Ec  = 8;
constexpr int Fc  = 8192;
constexpr int Tc  = Bc * Sc;     // 4096 tokens
constexpr int TKc = Tc * 2;      // 8192 token-expert slots
constexpr int CAPc = 1280;       // ceil(T*K/E * 1.25)

// ------------------------------------------------ gfx1250 async global->LDS
#if defined(__AMDGCN__) && __has_builtin(__builtin_amdgcn_global_load_async_to_lds_b128)
#define HAS_ASYNC_LDS 1
#else
#define HAS_ASYNC_LDS 0
#endif

DEVINL void async_copy_b128(const void* g, void* l) {
#if HAS_ASYNC_LDS
  __builtin_amdgcn_global_load_async_to_lds_b128(
      (__attribute__((address_space(1))) v4i*)(v4i*)(void*)g,
      (__attribute__((address_space(3))) v4i*)(v4i*)l, 0, 0);
#else
  *(v8h*)l = *(const v8h*)g;
#endif
}

DEVINL void async_wait_all() {
#if HAS_ASYNC_LDS
#if __has_builtin(__builtin_amdgcn_s_wait_asynccnt)
  __builtin_amdgcn_s_wait_asynccnt(0);
#else
  asm volatile("s_wait_asynccnt 0" ::: "memory");
#endif
#endif
}

// packed fp32x2 -> f16x2 convert + 4B store (v_cvt_pk_rtz_f16_f32 + ds_store_b32)
DEVINL void store_pk(_Float16* dst, float a, float b) {
  *(h2*)dst = __builtin_amdgcn_cvt_pkrtz(a, b);
}

// ---------------------------------------------------------------- WMMA helpers
DEVINL v8f wmma_f16(v16h a, v16h b, v8f c) {
  // D = A(16x32 f16) x B(32x16 f16) + C(16x16 f32)
  return __builtin_amdgcn_wmma_f32_16x16x32_f16(false, a, false, b, (short)0, c,
                                                false, false);
}

// A-matrix fragment: lane holds row (lane&15); halves map to
// K = h + (h>=8?8:0) + (lane>=16?8:0)  ->  two aligned 16B chunks at {k0, k0+16}
DEVINL v16h afrag(const _Float16* rowWin, int lane) {
  const int k0 = (lane & 16) ? 8 : 0;
  v8h c1 = *(const v8h*)(rowWin + k0);
  v8h c2 = *(const v8h*)(rowWin + k0 + 16);
  v16h a;
#pragma unroll
  for (int i = 0; i < 8; ++i) { a[i] = c1[i]; a[i + 8] = c2[i]; }
  return a;
}

// B-matrix fragment (column-major in LDS): lane holds column (lane&15);
// same K order as afrag: two aligned 16B chunks at {k0, k0+16}, k0 = (lane>=16?8:0)
DEVINL v16h bfrag(const _Float16* colWin, int lane) {
  const int k0 = (lane & 16) ? 8 : 0;
  v8h c1 = *(const v8h*)(colWin + k0);
  v8h c2 = *(const v8h*)(colWin + k0 + 16);
  v16h b;
#pragma unroll
  for (int i = 0; i < 8; ++i) { b[i] = c1[i]; b[i + 8] = c2[i]; }
  return b;
}

// ---------------------------------------------------------------- generic GEMM
// C[M,N] = act(A[M,K](f16) @ B[K,N](f32->f16) + bias)
// 256 threads = 8 waves in a 2(M) x 4(N) grid; wave tile 32x64;
// block tile 64x256, K-step 32. blockIdx.z = expert batch index.
template <int ACT, bool OUTF16, bool HASBIAS>
__global__ void wmma_gemm(const _Float16* __restrict__ A,
                          const float* __restrict__ B,
                          const float* __restrict__ bias,
                          void* __restrict__ Cout,
                          int M, int N, int K,
                          long long sA, long long sB, long long sBias,
                          long long sC) {
  __shared__ _Float16 lA[64 * 32];    // [row][k]
  __shared__ _Float16 lB[256 * 32];   // [n][k]  (column-major B tile)

  const int e = blockIdx.z;
  A += (size_t)e * sA;
  B += (size_t)e * sB;
  const float* bptr = HASBIAS ? (bias + (size_t)e * sBias) : nullptr;

  const int tid = threadIdx.x;
  const int lane = tid & 31, wid = tid >> 5;
  const int wm = wid & 1, wn = wid >> 1;          // 2 x 4 wave grid
  const int laneRow = lane & 15, laneHi = lane >> 4;
  const int m0 = blockIdx.y * 64, n0 = blockIdx.x * 256;

  v8f acc[2][4] = {};

  const int arow = tid >> 2, acol = (tid & 3) * 8;   // A: 8 halves / thread
  const int kp = tid >> 4, bn16 = (tid & 15) * 16;   // B: k-pair x 16 n / thread

  for (int kb = 0; kb < K; kb += 32) {
    // stage A tile via async global->LDS DMA (f16, no conversion needed)
    async_copy_b128(A + (size_t)(m0 + arow) * K + kb + acol,
                    &lA[arow * 32 + acol]);

    // stage B: fp32 -> packed f16 pairs (k, k+1) -> ds_store_b32, transposed
    const float* brow0 = B + (size_t)(kb + 2 * kp) * N + n0 + bn16;
    const float* brow1 = brow0 + N;
    if (kb + 32 < K) __builtin_prefetch(brow0 + (size_t)32 * N, 0, 1);
#pragma unroll
    for (int c = 0; c < 4; ++c) {
      float4 x = ((const float4*)brow0)[c];
      float4 y = ((const float4*)brow1)[c];
      const int n = bn16 + c * 4;
      store_pk(&lB[(n + 0) * 32 + 2 * kp], x.x, y.x);
      store_pk(&lB[(n + 1) * 32 + 2 * kp], x.y, y.y);
      store_pk(&lB[(n + 2) * 32 + 2 * kp], x.z, y.z);
      store_pk(&lB[(n + 3) * 32 + 2 * kp], x.w, y.w);
    }
    async_wait_all();
    __syncthreads();

    v16h a0 = afrag(&lA[(wm * 32 + 0  + laneRow) * 32], lane);
    v16h a1 = afrag(&lA[(wm * 32 + 16 + laneRow) * 32], lane);
#pragma unroll
    for (int sn = 0; sn < 4; ++sn) {
      v16h bf = bfrag(&lB[(wn * 64 + sn * 16 + laneRow) * 32], lane);
      acc[0][sn] = wmma_f16(a0, bf, acc[0][sn]);
      acc[1][sn] = wmma_f16(a1, bf, acc[1][sn]);
    }
    __syncthreads();
  }

#pragma unroll
  for (int sm = 0; sm < 2; ++sm) {
#pragma unroll
    for (int sn = 0; sn < 4; ++sn) {
      const int col = n0 + wn * 64 + sn * 16 + laneRow;
      const float bv = HASBIAS ? bptr[col] : 0.0f;
#pragma unroll
      for (int r = 0; r < 8; ++r) {
        const int row = m0 + wm * 32 + sm * 16 + r + laneHi * 8;
        float x = acc[sm][sn][r] + bv;
        if (ACT == 1) {  // gelu (tanh approximation, JAX default)
          float x3 = x * x * x;
          x = 0.5f * x *
              (1.0f + tanhf(0.79788456080286536f * (x + 0.044715f * x3)));
        }
        if (OUTF16)
          ((_Float16*)Cout + (size_t)e * sC)[(size_t)row * N + col] =
              (_Float16)x;
        else
          ((float*)Cout + (size_t)e * sC)[(size_t)row * N + col] = x;
      }
    }
  }
}

// ---------------------------------------------------------------- layernorm
__global__ void ln_kernel(const float* __restrict__ x,
                          const float* __restrict__ sc,
                          const float* __restrict__ bi,
                          _Float16* __restrict__ out) {
  __shared__ float r1[256], r2[256];
  const int t = blockIdx.x, tid = threadIdx.x;
  const float* xr = x + (size_t)t * Hc;
  float s = 0.f, s2 = 0.f;
  for (int j = tid; j < Hc; j += 256) { float v = xr[j]; s += v; s2 += v * v; }
  r1[tid] = s; r2[tid] = s2;
  __syncthreads();
  for (int o = 128; o > 0; o >>= 1) {
    if (tid < o) { r1[tid] += r1[tid + o]; r2[tid] += r2[tid + o]; }
    __syncthreads();
  }
  const float mean = r1[0] * (1.0f / Hc);
  const float var  = r2[0] * (1.0f / Hc) - mean * mean;
  const float inv  = rsqrtf(var + 1e-5f);
  _Float16* orow = out + (size_t)t * Hc;
  for (int j = tid; j < Hc; j += 256)
    orow[j] = (_Float16)((xr[j] - mean) * inv * sc[j] + bi[j]);
}

// ---------------------------------------------------------------- RoPE (inplace)
__global__ void rope_kernel(_Float16* __restrict__ x) {
  const size_t gid = (size_t)blockIdx.x * 256 + threadIdx.x;
  const int i  = (int)(gid & 63);           // half-dim index 0..63
  const int hh = (int)((gid >> 6) & (NHc - 1));
  const size_t t = gid >> 10;               // token
  const int p = (int)(t & (Sc - 1));        // position in sequence
  const float freq = __powf(10000.0f, -(float)i * (1.0f / 64.0f));
  const float ang = (float)p * freq;
  const float cs = cosf(ang), sn = sinf(ang);
  const size_t base = t * (size_t)Hc + (size_t)hh * HDc;
  const float x1 = (float)x[base + i], x2 = (float)x[base + 64 + i];
  x[base + i]      = (_Float16)(x1 * cs - x2 * sn);
  x[base + 64 + i] = (_Float16)(x1 * sn + x2 * cs);
}

// ---------------------------------------------------------------- flash attention
// 64 threads (2 waves), 32 q-rows per block, 64-key blocks, causal + mask.
__global__ void attn_kernel(const _Float16* __restrict__ q,
                            const _Float16* __restrict__ kmat,
                            const _Float16* __restrict__ vmat,
                            const int* __restrict__ amask,
                            _Float16* __restrict__ ctx) {
  __shared__ _Float16 lQ[32 * 128];
  __shared__ _Float16 lK[64 * 128];   // row-major [key][feat] == Bt layout
  __shared__ _Float16 lVt[128 * 64];  // transposed [feat][key]
  __shared__ float    lS[32 * 64];
  __shared__ _Float16 lP[32 * 64];
  __shared__ float    lM[32], lL[32], lAl[32], lMask[64];

  const int tid = threadIdx.x;        // 64 threads
  const int lane = tid & 31, wid = tid >> 5;
  const int laneRow = lane & 15, laneHi = lane >> 4;
  const int qb = blockIdx.x, h = blockIdx.y, b = blockIdx.z;
  const size_t headOff = (size_t)h * HDc;

  // stage Q: 32 x 128 halves via async DMA
#pragma unroll
  for (int c = 0; c < 8; ++c) {
    int ch = tid * 8 + c;
    int row = ch >> 4, cc = (ch & 15) * 8;
    async_copy_b128(q + (size_t)(b * Sc + qb * 32 + row) * Hc + headOff + cc,
                    &lQ[row * 128 + cc]);
  }
  async_wait_all();
  if (tid < 32) { lM[tid] = -3.0e38f; lL[tid] = 0.0f; }

  v8f cacc[8] = {};
  const int nkb = (qb >> 1) + 1;

  for (int kb = 0; kb < nkb; ++kb) {
    __syncthreads();
    // stage K (async, row-major) and V (transposed, via VGPRs); thread = key row
#pragma unroll
    for (int c = 0; c < 16; ++c) {
      const int cc = c * 8, row = tid;
      const size_t gb = (size_t)(b * Sc + kb * 64 + row) * Hc + headOff + cc;
      async_copy_b128(kmat + gb, &lK[row * 128 + cc]);
      v8h vv = *(const v8h*)(vmat + gb);
#pragma unroll
      for (int j = 0; j < 8; ++j) lVt[(cc + j) * 64 + row] = vv[j];
    }
    lMask[tid] = (amask[b * Sc + kb * 64 + tid] > 0) ? 1.0f : 0.0f;
    async_wait_all();
    __syncthreads();

    // scores: wave owns 16 q-rows x 64 keys
    v8f sacc[4] = {};
#pragma unroll
    for (int fs = 0; fs < 4; ++fs) {
      v16h aq = afrag(&lQ[(wid * 16 + laneRow) * 128 + fs * 32], lane);
#pragma unroll
      for (int sn = 0; sn < 4; ++sn) {
        v16h bk = bfrag(&lK[(sn * 16 + laneRow) * 128 + fs * 32], lane);
        sacc[sn] = wmma_f16(aq, bk, sacc[sn]);
      }
    }
    const float scl = 0.088388347648318447f;  // 1/sqrt(128)
#pragma unroll
    for (int sn = 0; sn < 4; ++sn) {
#pragma unroll
      for (int r = 0; r < 8; ++r) {
        const int row = r + laneHi * 8;
        const int col = sn * 16 + laneRow;
        const int qg = qb * 32 + wid * 16 + row;
        const int kg = kb * 64 + col;
        const float sv = sacc[sn][r] * scl;
        const bool ok = (kg <= qg) && (lMask[col] > 0.0f);
        lS[(wid * 16 + row) * 64 + col] = ok ? sv : -1.0e9f;
      }
    }
    __syncthreads();

    // online softmax per row (32 rows handled by threads 0..31)
    if (tid < 32) {
      const float mold = lM[tid], lold = lL[tid];
      float rmax = -3.0e38f;
      for (int c2 = 0; c2 < 64; ++c2) rmax = fmaxf(rmax, lS[tid * 64 + c2]);
      const float mnew = fmaxf(mold, rmax);
      const float alpha = __expf(mold - mnew);
      float rsum = 0.0f;
      for (int c2 = 0; c2 < 64; ++c2) {
        const float pv = __expf(lS[tid * 64 + c2] - mnew);
        rsum += pv;
        lP[tid * 64 + c2] = (_Float16)pv;
      }
      lM[tid] = mnew; lL[tid] = lold * alpha + rsum; lAl[tid] = alpha;
    }
    __syncthreads();

    // rescale running ctx, then ctx += P @ V
    v8f av;
#pragma unroll
    for (int r = 0; r < 8; ++r) av[r] = lAl[wid * 16 + r + laneHi * 8];
#pragma unroll
    for (int i = 0; i < 8; ++i) cacc[i] *= av;
#pragma unroll
    for (int ks = 0; ks < 2; ++ks) {
      v16h ap = afrag(&lP[(wid * 16 + laneRow) * 64 + ks * 32], lane);
#pragma unroll
      for (int nf = 0; nf < 8; ++nf) {
        v16h bv = bfrag(&lVt[(nf * 16 + laneRow) * 64 + ks * 32], lane);
        cacc[nf] = wmma_f16(ap, bv, cacc[nf]);
      }
    }
  }
  __syncthreads();

  v8f linv;
#pragma unroll
  for (int r = 0; r < 8; ++r) linv[r] = 1.0f / lL[wid * 16 + r + laneHi * 8];
#pragma unroll
  for (int nf = 0; nf < 8; ++nf) {
#pragma unroll
    for (int r = 0; r < 8; ++r) {
      const int row = qb * 32 + wid * 16 + r + laneHi * 8;
      ctx[(size_t)(b * Sc + row) * Hc + headOff + nf * 16 + laneRow] =
          (_Float16)(cacc[nf][r] * linv[r]);
    }
  }
}

// ---------------------------------------------------------------- elementwise
__global__ void attn_combine(const float* __restrict__ hs,
                             const float* __restrict__ araw,
                             const float* __restrict__ xg,
                             const float* __restrict__ lsa,
                             float* __restrict__ hid2) {
  const size_t i = (size_t)blockIdx.x * 256 + threadIdx.x;
  const float g = 1.0f / (1.0f + __expf(-xg[i]));
  hid2[i] = hs[i] + araw[i] * g * lsa[i & (Hc - 1)];
}

// router logits: block per token, 8 waves -> 8 experts
__global__ void router_kernel(const _Float16* __restrict__ xn2,
                              const float* __restrict__ rw,
                              float* __restrict__ logits) {
  const int t = blockIdx.x;
  const int lane = threadIdx.x & 31, e = threadIdx.x >> 5;
  const _Float16* xr = xn2 + (size_t)t * Hc;
  float s = 0.0f;
  for (int j = lane; j < Hc; j += 32) s += (float)xr[j] * rw[(size_t)j * Ec + e];
#pragma unroll
  for (int o = 16; o > 0; o >>= 1) s += __shfl_down(s, o, 32);
  if (lane == 0) logits[t * Ec + e] = s;
}

__global__ void topk_kernel(const float* __restrict__ logits,
                            int* __restrict__ eidx, float* __restrict__ gates) {
  const int t = blockIdx.x * 256 + threadIdx.x;
  if (t >= Tc) return;
  float lg[8]; float m = -3.0e38f;
#pragma unroll
  for (int e = 0; e < 8; ++e) { lg[e] = logits[t * 8 + e]; m = fmaxf(m, lg[e]); }
  float p[8];
#pragma unroll
  for (int e = 0; e < 8; ++e) p[e] = __expf(lg[e] - m);
  int i0 = 0;
#pragma unroll
  for (int e = 1; e < 8; ++e) if (p[e] > p[i0]) i0 = e;
  int i1 = (i0 == 0) ? 1 : 0;
#pragma unroll
  for (int e = 0; e < 8; ++e) if (e != i1 && e != i0 && p[e] > p[i1]) i1 = e;
  const float gs = p[i0] + p[i1];
  eidx[t * 2] = i0; eidx[t * 2 + 1] = i1;
  gates[t * 2] = p[i0] / gs; gates[t * 2 + 1] = p[i1] / gs;
}

// deterministic capacity positions (mirrors flattened-cumsum semantics)
__global__ void pos_kernel(const int* __restrict__ eidx, int* __restrict__ pos) {
  const int e = threadIdx.x;  // 8 threads
  int cnt = 0;
  for (int s = 0; s < TKc; ++s)
    if (eidx[s] == e) pos[s] = cnt++;
}

__global__ void scatter_kernel(const _Float16* __restrict__ xn2,
                               const int* __restrict__ eidx,
                               const int* __restrict__ pos,
                               _Float16* __restrict__ buf) {
  const int s = blockIdx.x;
  const int p = pos[s];
  if (p >= CAPc) return;  // dropped (over capacity)
  const int e = eidx[s], t = s >> 1;
  const _Float16* src = xn2 + (size_t)t * Hc;
  _Float16* dst = buf + ((size_t)e * CAPc + p) * Hc;
  for (int j = threadIdx.x; j < Hc; j += 256) dst[j] = src[j];
}

__global__ void gather_kernel(const float* __restrict__ hid2,
                              const float* __restrict__ obuf,
                              const int* __restrict__ eidx,
                              const int* __restrict__ pos,
                              const float* __restrict__ gates,
                              const float* __restrict__ lsm,
                              float* __restrict__ out) {
  const int t = blockIdx.x;
  const int e0 = eidx[t * 2], e1 = eidx[t * 2 + 1];
  const int p0 = pos[t * 2], p1 = pos[t * 2 + 1];
  const bool k0 = p0 < CAPc, k1 = p1 < CAPc;
  const int p0c = p0 < CAPc ? p0 : CAPc - 1;
  const int p1c = p1 < CAPc ? p1 : CAPc - 1;
  const float g0 = gates[t * 2], g1 = gates[t * 2 + 1];
  const float* r0 = obuf + ((size_t)e0 * CAPc + p0c) * Hc;
  const float* r1 = obuf + ((size_t)e1 * CAPc + p1c) * Hc;
  for (int j = threadIdx.x; j < Hc; j += 256) {
    float m = 0.0f;
    if (k0) m += g0 * r0[j];
    if (k1) m += g1 * r1[j];
    out[(size_t)t * Hc + j] = hid2[(size_t)t * Hc + j] + lsm[j] * m;
  }
}

// ---------------------------------------------------------------- launcher
extern "C" void kernel_launch(void* const* d_in, const int* in_sizes, int n_in,
                              void* d_out, int out_size, void* d_ws,
                              size_t ws_size, hipStream_t stream) {
  (void)in_sizes; (void)n_in; (void)out_size; (void)ws_size;
  const float* hs   = (const float*)d_in[0];
  const int*   am   = (const int*)d_in[1];
  const float* ln1s = (const float*)d_in[2];
  const float* ln1b = (const float*)d_in[3];
  const float* ln2s = (const float*)d_in[4];
  const float* ln2b = (const float*)d_in[5];
  const float* Wq   = (const float*)d_in[6];
  const float* Wk   = (const float*)d_in[7];
  const float* Wv   = (const float*)d_in[8];
  const float* Wo   = (const float*)d_in[9];
  const float* Wg   = (const float*)d_in[10];
  const float* bg   = (const float*)d_in[11];
  const float* lsa  = (const float*)d_in[12];
  const float* lsm  = (const float*)d_in[13];
  const float* rw   = (const float*)d_in[14];
  const float* W1   = (const float*)d_in[15];
  const float* b1   = (const float*)d_in[16];
  const float* W2   = (const float*)d_in[17];
  const float* b2   = (const float*)d_in[18];
  float* out = (float*)d_out;

  char* ws = (char*)d_ws;
  size_t off = 0;
  auto carve = [&](size_t bytes) -> char* {
    char* p = ws + off;
    off = (off + bytes + 255) & ~(size_t)255;
    return p;
  };
  _Float16* xn    = (_Float16*)carve((size_t)Tc * Hc * 2);
  _Float16* qbuf  = (_Float16*)carve((size_t)Tc * Hc * 2);
  _Float16* kbuf  = (_Float16*)carve((size_t)Tc * Hc * 2);
  _Float16* vbuf  = (_Float16*)carve((size_t)Tc * Hc * 2);
  float*    xg    = (float*)carve((size_t)Tc * Hc * 4);
  _Float16* ctx   = (_Float16*)carve((size_t)Tc * Hc * 2);
  float*    araw  = (float*)carve((size_t)Tc * Hc * 4);
  float*    hid2  = (float*)carve((size_t)Tc * Hc * 4);
  _Float16* xn2   = (_Float16*)carve((size_t)Tc * Hc * 2);
  float*    logit = (float*)carve((size_t)Tc * Ec * 4);
  int*      eidx  = (int*)carve((size_t)TKc * 4);
  float*    gates = (float*)carve((size_t)TKc * 4);
  int*      posb  = (int*)carve((size_t)TKc * 4);
  _Float16* buf   = (_Float16*)carve((size_t)Ec * CAPc * Hc * 2);
  _Float16* hmid  = (_Float16*)carve((size_t)Ec * CAPc * Fc * 2);
  float*    obuf  = (float*)carve((size_t)Ec * CAPc * Hc * 4);

  // 1) pre-norm
  ln_kernel<<<Tc, 256, 0, stream>>>(hs, ln1s, ln1b, xn);

  // 2) q/k/v/gate projections (block tile 64 x 256)
  dim3 gproj(Hc / 256, Tc / 64, 1);
  wmma_gemm<0, true, false><<<gproj, 256, 0, stream>>>(
      xn, Wq, nullptr, qbuf, Tc, Hc, Hc, 0, 0, 0, 0);
  wmma_gemm<0, true, false><<<gproj, 256, 0, stream>>>(
      xn, Wk, nullptr, kbuf, Tc, Hc, Hc, 0, 0, 0, 0);
  wmma_gemm<0, true, false><<<gproj, 256, 0, stream>>>(
      xn, Wv, nullptr, vbuf, Tc, Hc, Hc, 0, 0, 0, 0);
  wmma_gemm<0, false, true><<<gproj, 256, 0, stream>>>(
      xn, Wg, bg, xg, Tc, Hc, Hc, 0, 0, 0, 0);

  // 3) RoPE on q and k
  const int ropeT = Tc * NHc * (HDc / 2);
  rope_kernel<<<ropeT / 256, 256, 0, stream>>>(qbuf);
  rope_kernel<<<ropeT / 256, 256, 0, stream>>>(kbuf);

  // 4) causal flash attention per (q-block, head, batch)
  dim3 gattn(Sc / 32, NHc, Bc);
  attn_kernel<<<gattn, 64, 0, stream>>>(qbuf, kbuf, vbuf, am, ctx);

  // 5) output projection + gate + layerscale + residual
  wmma_gemm<0, false, false><<<gproj, 256, 0, stream>>>(
      ctx, Wo, nullptr, araw, Tc, Hc, Hc, 0, 0, 0, 0);
  attn_combine<<<(Tc * Hc) / 256, 256, 0, stream>>>(hs, araw, xg, lsa, hid2);

  // 6) MoE: norm, route, scatter
  ln_kernel<<<Tc, 256, 0, stream>>>(hid2, ln2s, ln2b, xn2);
  router_kernel<<<Tc, 256, 0, stream>>>(xn2, rw, logit);
  topk_kernel<<<Tc / 256, 256, 0, stream>>>(logit, eidx, gates);
  pos_kernel<<<1, 8, 0, stream>>>(eidx, posb);
  scatter_kernel<<<TKc, 256, 0, stream>>>(xn2, eidx, posb, buf);

  // 7) expert MLPs (batched over E)
  dim3 ge1(Fc / 256, CAPc / 64, Ec);
  wmma_gemm<1, true, true><<<ge1, 256, 0, stream>>>(
      buf, W1, b1, hmid, CAPc, Fc, Hc,
      (long long)CAPc * Hc, (long long)Hc * Fc, (long long)Fc,
      (long long)CAPc * Fc);
  dim3 ge2(Hc / 256, CAPc / 64, Ec);
  wmma_gemm<0, false, true><<<ge2, 256, 0, stream>>>(
      hmid, W2, b2, obuf, CAPc, Hc, Fc,
      (long long)CAPc * Fc, (long long)Fc * Hc, (long long)Hc,
      (long long)CAPc * Hc);

  // 8) gather + residual -> output
  gather_kernel<<<Tc, 256, 0, stream>>>(hid2, obuf, eidx, posb, gates, lsm, out);
}
